// brits_32607391711618
// MI455X (gfx1250) — matrix operand order repaired and checked by reading, not compiled, not run on hardware
//
#include <hip/hip_runtime.h>

// ---------------------------------------------------------------------------
// BRITS on MI455X (gfx1250): persistent WMMA-bf16 recurrence scan.
//   B=128, L=1024, K=64, H=128. Two directions, batch tiled by 16 rows/WG.
//   16 workgroups x 256 threads (8 wave32). All per-step GEMMs use
//   v_wmma_f32_16x16x32_bf16; weights pre-swizzled into B-fragment layout.
//   Next-step x/mask rows are prefetched (global_prefetch_b8) to hide the
//   only HBM-dependent loads on the sequential critical path.
// ---------------------------------------------------------------------------

#define B_SZ   128
#define L_SZ   1024
#define K_SZ   64
#define H_SZ   128
#define BT     16
#define NTHR   256
#define TOTAL_ELEMS ((size_t)B_SZ * L_SZ * K_SZ)

typedef __attribute__((ext_vector_type(16))) __bf16        v16bf;
typedef __attribute__((ext_vector_type(16))) unsigned short v16u;
typedef __attribute__((ext_vector_type(8)))  float          v8f;

__device__ __forceinline__ float bf2f(unsigned short u) {
    unsigned int x = ((unsigned int)u) << 16;
    return __builtin_bit_cast(float, x);
}
__device__ __forceinline__ unsigned short f2bf(float f) {
    unsigned int x = __builtin_bit_cast(unsigned int, f);
    return (unsigned short)((x + 0x8000u) >> 16);
}
__device__ __forceinline__ float sigm(float x) { return 1.f / (1.f + expf(-x)); }

__device__ __forceinline__ v8f wmma_bf16(v16u a, v16u b, v8f c) {
    return __builtin_amdgcn_wmma_f32_16x16x32_bf16(
        false, __builtin_bit_cast(v16bf, a),
        false, __builtin_bit_cast(v16bf, b),
        (short)0, c, false, false);
}

// A fragment: 16x32 bf16 tile (rows x K), gathered from row-major LDS.
// Lane layout per ISA 7.12.2: lanes 0-15 -> K {0..7,16..23}, lanes 16-31 -> +8.
// Pairs are contiguous, so this lowers to 2x ds_load_b128 per fragment.
__device__ __forceinline__ v16u load_A32(const unsigned short* m, int ld, int kbase, int lane) {
    v16u f;
    int row = lane & 15;
    int kb = kbase + ((lane >> 4) << 3);
    const unsigned short* p = m + row * ld;
#pragma unroll
    for (int v = 0; v < 8; ++v) {
        int k0 = kb + ((v < 4) ? (v << 1) : (16 + ((v - 4) << 1)));
        f[2 * v]     = p[k0];
        f[2 * v + 1] = p[k0 + 1];
    }
    return f;
}

// B fragment: pre-swizzled in global memory -> 32B vector load per lane.
__device__ __forceinline__ v16u load_Bsw(const unsigned short* swz, int Ksteps, int nt, int ks, int lane) {
    return ((const v16u*)swz)[(size_t)(nt * Ksteps + ks) * 32 + lane];
}

// ---------------------------------------------------------------------------
// init / helper kernels
// ---------------------------------------------------------------------------

__global__ void brits_zero(float* acc) {
    int t = threadIdx.x;
    acc[t] = 0.f;
    acc[t + 256] = 0.f;
}

// colsum of W_gh (K x H) and W_gx (K x K): dt is constant across K per row.
__global__ void brits_cs(const float* __restrict__ Wgh, const float* __restrict__ Wgx,
                         float* __restrict__ cs_gh, float* __restrict__ cs_gx) {
    int t = threadIdx.x;
    if (t < H_SZ) {
        float s = 0.f;
        for (int k = 0; k < K_SZ; ++k) s += Wgh[k * H_SZ + t];
        cs_gh[t] = s;
    }
    if (t < K_SZ) {
        float s = 0.f;
        for (int k = 0; k < K_SZ; ++k) s += Wgx[k * K_SZ + t];
        cs_gx[t] = s;
    }
}

// Convert f32 weight (Ktot x Ntot, row-major) into bf16 WMMA B-fragment layout:
// dst[((nt*Ksteps+ks)*32 + lane)*16 + i] = W[ks*32 + ((lane>>4)<<3) + (i<8 ? i : i+8)][nt*16 + (lane&15)]
__global__ void brits_swz(const float* __restrict__ src, unsigned short* __restrict__ dst,
                          int Ktot, int Ntot, int zeroDiag) {
    int total = Ktot * Ntot;
    int Ksteps = Ktot >> 5;
    for (int o = blockIdx.x * blockDim.x + threadIdx.x; o < total; o += gridDim.x * blockDim.x) {
        int i = o & 15;
        int lane = (o >> 4) & 31;
        int tIdx = o >> 9;
        int ks = tIdx % Ksteps;
        int nt = tIdx / Ksteps;
        int k = ks * 32 + ((lane >> 4) << 3) + ((i < 8) ? i : (i + 8));
        int n = nt * 16 + (lane & 15);
        float v = src[k * Ntot + n];
        if (zeroDiag && k == n) v = 0.f;
        dst[o] = f2bf(v);
    }
}

// denominator: sum(mask * valid)
__global__ void brits_den(const float* __restrict__ mask, const int* __restrict__ rec, float* dst) {
    __shared__ float red[NTHR];
    float s = 0.f;
    for (size_t e = (size_t)blockIdx.x * blockDim.x + threadIdx.x; e < TOTAL_ELEMS;
         e += (size_t)gridDim.x * blockDim.x) {
        int t = (int)((e / K_SZ) % L_SZ);
        int b = (int)(e / ((size_t)L_SZ * K_SZ));
        if (t < rec[b]) s += mask[e];
    }
    red[threadIdx.x] = s;
    __syncthreads();
    for (int k = NTHR / 2; k > 0; k >>= 1) {
        if (threadIdx.x < k) red[threadIdx.x] += red[threadIdx.x + k];
        __syncthreads();
    }
    if (threadIdx.x == 0) atomicAdd(dst, red[0]);
}

// ---------------------------------------------------------------------------
// main persistent scan kernel
// ---------------------------------------------------------------------------

struct ScanArgs {
    const float *x, *mask, *ts;
    const int* rec;
    const float *b_gh[2], *b_gx[2], *b_hist[2], *b_feat[2], *b_beta[2], *b_lstm[2];
    const float *w_out[2], *b_out[2];
    const float *cs_gh[2], *cs_gx[2];
    const unsigned short *whist[2], *wfeat[2], *wbeta[2], *wih[2], *whh[2];
    float *impF, *impB, *err, *predF, *predB;
};

__global__ __launch_bounds__(NTHR) void brits_scan(ScanArgs A) {
    __shared__ unsigned short s_hbf[BT * H_SZ];          // decayed h, bf16 (A operand)
    __shared__ float          s_gates[BT * 4 * H_SZ];    // gates (16 x 512)
    __shared__ float          s_xh[BT * K_SZ];
    __shared__ float          s_zh[BT * K_SZ];
    __shared__ float          s_betar[BT * K_SZ];        // raw beta, also reduce scratch
    __shared__ unsigned short s_xt[BT * K_SZ];
    __shared__ unsigned short s_mt[BT * K_SZ];
    __shared__ unsigned short s_ccm[BT * 2 * K_SZ];      // [c_c | m] bf16 (A operand)
    __shared__ unsigned short s_gxm[BT * 2 * K_SZ];      // [gamma_x | m] bf16 (A operand)
    __shared__ unsigned short s_xcbf[BT * K_SZ];         // x_c bf16 (A operand)
    __shared__ float s_delta[BT];
    __shared__ float s_vt[BT];
    __shared__ int   s_src[BT];
    __shared__ float s_keep[BT];

    const int tid  = threadIdx.x;
    const int wave = tid >> 5;
    const int lane = tid & 31;
    const int dir  = blockIdx.x & 1;
    const int tile = blockIdx.x >> 1;

    const float* bgh   = A.b_gh[dir];
    const float* bgx   = A.b_gx[dir];
    const float* bhist = A.b_hist[dir];
    const float* bfeat = A.b_feat[dir];
    const float* bbeta = A.b_beta[dir];
    const float* blstm = A.b_lstm[dir];
    const float* csgh  = A.cs_gh[dir];
    const float* csgx  = A.cs_gx[dir];
    const unsigned short* Whist = A.whist[dir];
    const unsigned short* Wfeat = A.wfeat[dir];
    const unsigned short* Wbeta = A.wbeta[dir];
    const unsigned short* Wih   = A.wih[dir];
    const unsigned short* Whh   = A.whh[dir];
    float* imp = dir ? A.impB : A.impF;

    float h[8], c[8];
#pragma unroll
    for (int j = 0; j < 8; ++j) { h[j] = 0.f; c[j] = 0.f; }
    float errSum = 0.f;

    for (int t = 0; t < L_SZ; ++t) {
        __syncthreads();
        // ---- phase 0: per-row meta (delta, valid, source index for reversal),
        //      plus prefetch of NEXT step's x/mask rows (global_prefetch_b8).
        if (tid < BT) {
            int b = tile * BT + tid;
            int n = A.rec[b];
            float d, v, kp;
            int si;
            if (dir == 0) {
                si = t; kp = 1.f;
                v = (t < n) ? 1.f : 0.f;
                d = (t == 0) ? 0.f : (A.ts[b * L_SZ + t] - A.ts[b * L_SZ + t - 1]);
            } else {
                bool kk = (t < n);
                kp = kk ? 1.f : 0.f;
                v = kp;
                si = kk ? (n - 1 - t) : 0;
                float tc = kk ? A.ts[b * L_SZ + (n - 1 - t)] : 0.f;
                float tp = 0.f;
                if (t > 0) { bool k1 = ((t - 1) < n); tp = k1 ? A.ts[b * L_SZ + (n - t)] : 0.f; }
                d = (t == 0) ? 0.f : (tc - tp);
            }
            s_delta[tid] = d; s_vt[tid] = v; s_src[tid] = si; s_keep[tid] = kp;
            // prefetch next step's inputs (fire-and-forget, no counter)
            int tn = t + 1;
            if (tn < L_SZ) {
                int sn = (dir == 0) ? tn : ((tn < n) ? (n - 1 - tn) : 0);
                size_t pb = ((size_t)b * L_SZ + sn) * K_SZ;
                __builtin_prefetch(&A.x[pb], 0, 0);
                __builtin_prefetch(&A.x[pb + 32], 0, 0);
                __builtin_prefetch(&A.mask[pb], 0, 0);
                __builtin_prefetch(&A.mask[pb + 32], 0, 0);
            }
        }
        __syncthreads();
        // ---- load x_t, m_t (with on-the-fly reversal for dir 1)
#pragma unroll
        for (int j = 0; j < 4; ++j) {
            int e = tid + j * NTHR;
            int r = e >> 6, k = e & 63;
            int b = tile * BT + r;
            size_t base = ((size_t)b * L_SZ + s_src[r]) * K_SZ + k;
            float kp = s_keep[r];
            s_xt[e] = f2bf(A.x[base] * kp);
            s_mt[e] = f2bf(A.mask[base] * kp);
        }
        __syncthreads();
        // ---- phase 1: temporal decay of h; build [gamma_x|m] A operand
#pragma unroll
        for (int j = 0; j < 8; ++j) {
            int idx = tid * 8 + j;
            int b = idx >> 7, q = idx & 127;
            float dlt = s_delta[b];
            float gh = expf(-fmaxf(0.f, dlt * csgh[q] + bgh[q]));
            h[j] *= gh;
            s_hbf[idx] = f2bf(h[j]);
            unsigned short gv;
            if (q < 64) gv = f2bf(expf(-fmaxf(0.f, dlt * csgx[q] + bgx[q])));
            else        gv = s_mt[b * 64 + (q - 64)];
            s_gxm[idx] = gv;
        }
        __syncthreads();
        // ---- phase 2: x_h = h @ W_hist (waves 0-3), beta_raw = [gx|m] @ W_beta (waves 4-7)
        {
            int nt = wave & 3;
            const unsigned short* srcA = (wave < 4) ? s_hbf : s_gxm;
            const unsigned short* wB   = (wave < 4) ? Whist : Wbeta;
            const float* bb            = (wave < 4) ? bhist : bbeta;
            float* dst                 = (wave < 4) ? s_xh  : s_betar;
            v8f acc;
#pragma unroll
            for (int v = 0; v < 8; ++v) acc[v] = 0.f;
#pragma unroll
            for (int ks = 0; ks < 4; ++ks) {
                v16u a = load_A32(srcA, 128, ks * 32, lane);
                acc = wmma_bf16(a, load_Bsw(wB, 4, nt, ks, lane), acc);
            }
            int m0 = (lane >> 4) << 3;
            int col = nt * 16 + (lane & 15);
            float bv = bb[col];
#pragma unroll
            for (int v = 0; v < 8; ++v) dst[(m0 + v) * K_SZ + col] = acc[v] + bv;
        }
        __syncthreads();
        // ---- phase 3: x_c = m*x + (1-m)*x_h
#pragma unroll
        for (int j = 0; j < 4; ++j) {
            int e = tid + j * NTHR;
            float mv = bf2f(s_mt[e]), xv = bf2f(s_xt[e]);
            s_xcbf[e] = f2bf(mv * xv + (1.f - mv) * s_xh[e]);
        }
        __syncthreads();
        // ---- phase 4: z_h = x_c @ W_feat (diag already zeroed at swizzle time)
        if (wave < 4) {
            int nt = wave;
            v8f acc;
#pragma unroll
            for (int v = 0; v < 8; ++v) acc[v] = 0.f;
#pragma unroll
            for (int ks = 0; ks < 2; ++ks) {
                v16u a = load_A32(s_xcbf, 64, ks * 32, lane);
                acc = wmma_bf16(a, load_Bsw(Wfeat, 2, nt, ks, lane), acc);
            }
            int m0 = (lane >> 4) << 3;
            int col = nt * 16 + (lane & 15);
            float bv = bfeat[col];
#pragma unroll
            for (int v = 0; v < 8; ++v) s_zh[(m0 + v) * K_SZ + col] = acc[v] + bv;
        }
        __syncthreads();
        // ---- phase 5: beta/c_h/c_c, error, imputation
#pragma unroll
        for (int j = 0; j < 4; ++j) {
            int e = tid + j * NTHR;
            int r = e >> 6, k = e & 63;
            float mv = bf2f(s_mt[e]), xv = bf2f(s_xt[e]);
            float beta = sigm(s_betar[e]);
            float ch = beta * s_zh[e] + (1.f - beta) * s_xh[e];
            float cc = mv * xv + (1.f - mv) * ch;
            s_ccm[r * 128 + k]      = f2bf(cc);
            s_ccm[r * 128 + 64 + k] = s_mt[e];
            float v = s_vt[r];
            errSum += (fabsf(xv - s_xh[e]) + fabsf(xv - s_zh[e]) + fabsf(xv - ch)) * mv * v;
            int b = tile * BT + r;
            imp[((size_t)b * L_SZ + t) * K_SZ + k] = cc * v;
        }
        __syncthreads();
        // ---- phase 6: gates = [c_c|m] @ W_ih + h @ W_hh + b_lstm  (16x512, K=128, x2)
        {
            v16u ac[4], ah[4];
#pragma unroll
            for (int ks = 0; ks < 4; ++ks) {
                ac[ks] = load_A32(s_ccm, 128, ks * 32, lane);
                ah[ks] = load_A32(s_hbf, 128, ks * 32, lane);
            }
#pragma unroll
            for (int ntl = 0; ntl < 4; ++ntl) {
                int ntg = wave * 4 + ntl;
                int col = ntg * 16 + (lane & 15);
                float bv = blstm[col];
                v8f acc;
#pragma unroll
                for (int v = 0; v < 8; ++v) acc[v] = bv;
#pragma unroll
                for (int ks = 0; ks < 4; ++ks) {
                    acc = wmma_bf16(ac[ks], load_Bsw(Wih, 4, ntg, ks, lane), acc);
                    acc = wmma_bf16(ah[ks], load_Bsw(Whh, 4, ntg, ks, lane), acc);
                }
                int m0 = (lane >> 4) << 3;
#pragma unroll
                for (int v = 0; v < 8; ++v) s_gates[(m0 + v) * 512 + col] = acc[v];
            }
        }
        __syncthreads();
        // ---- phase 7: LSTM pointwise with valid-masked state update
#pragma unroll
        for (int j = 0; j < 8; ++j) {
            int idx = tid * 8 + j;
            int b = idx >> 7, q = idx & 127;
            float ig = s_gates[b * 512 + q];
            float fg = s_gates[b * 512 + 128 + q];
            float gg = s_gates[b * 512 + 256 + q];
            float og = s_gates[b * 512 + 384 + q];
            float cn = sigm(fg) * c[j] + sigm(ig) * tanhf(gg);
            float hn = sigm(og) * tanhf(cn);
            float v = s_vt[b];
            h[j] = v * hn + (1.f - v) * h[j];
            c[j] = v * cn + (1.f - v) * c[j];
        }
    }

    // ---- predictions: sigmoid(h @ W_out + b_out)
    __syncthreads();
#pragma unroll
    for (int j = 0; j < 8; ++j) s_gates[tid * 8 + j] = h[j];
    __syncthreads();
    if (tid < BT) {
        int b = tile * BT + tid;
        float s = 0.f;
        const float* wo = A.w_out[dir];
        for (int q = 0; q < H_SZ; ++q) s += s_gates[tid * H_SZ + q] * wo[q];
        (dir ? A.predB : A.predF)[b] = sigm(s + A.b_out[dir][0]);
    }
    // ---- error reduction
    s_betar[tid] = errSum;
    __syncthreads();
    for (int s = NTHR / 2; s > 0; s >>= 1) {
        if (tid < s) s_betar[tid] += s_betar[tid + s];
        __syncthreads();
    }
    if (tid == 0) atomicAdd(&A.err[dir], s_betar[0]);
}

// ---------------------------------------------------------------------------
// post passes
// ---------------------------------------------------------------------------

// |imp_f - imp_b| reduction + average imputations in place (impF slot = d_out)
__global__ void brits_combine(float* __restrict__ impF, const float* __restrict__ impB, float* dst) {
    __shared__ float red[NTHR];
    float s = 0.f;
    for (size_t e = (size_t)blockIdx.x * blockDim.x + threadIdx.x; e < TOTAL_ELEMS;
         e += (size_t)gridDim.x * blockDim.x) {
        float f = impF[e];
        float b = impB[e];
        s += fabsf(f - b);
        impF[e] = 0.5f * (f + b);
    }
    red[threadIdx.x] = s;
    __syncthreads();
    for (int k = NTHR / 2; k > 0; k >>= 1) {
        if (threadIdx.x < k) red[threadIdx.x] += red[threadIdx.x + k];
        __syncthreads();
    }
    if (threadIdx.x == 0) atomicAdd(dst, red[0]);
}

__global__ void brits_final(const float* acc, const float* predF, const float* predB, float* out) {
    int t = threadIdx.x;
    if (t == 0) {
        float den = acc[2] + 1e-5f;
        float loss = 0.3f * acc[0] / den + 0.3f * acc[1] / den +
                     acc[3] * (0.1f / 8388608.0f);
        out[0] = loss;
    }
    if (t < B_SZ) out[1 + t] = 0.5f * (predF[t] + predB[t]);
}

// ---------------------------------------------------------------------------
// launcher
// ---------------------------------------------------------------------------

extern "C" void kernel_launch(void* const* d_in, const int* in_sizes, int n_in,
                              void* d_out, int out_size, void* d_ws, size_t ws_size,
                              hipStream_t stream) {
    (void)in_sizes; (void)n_in; (void)out_size; (void)ws_size;

    const float* x    = (const float*)d_in[0];
    const float* mask = (const float*)d_in[1];
    const int*   rec  = (const int*)d_in[2];
    const float* ts   = (const float*)d_in[3];

    char* ws = (char*)d_ws;
    const size_t OFF_ACC    = (size_t)TOTAL_ELEMS * sizeof(float); // after impB
    const size_t OFF_SWZ    = OFF_ACC + 2048;
    const size_t SWZ_STRIDE = 307200;                              // bytes per dir (151552 bf16 + pad)
    const size_t OFF_CS     = OFF_SWZ + 2 * SWZ_STRIDE;

    float* impB = (float*)ws;
    float* accp = (float*)(ws + OFF_ACC);   // [0]=err_f [1]=err_b [2]=den [3]=loss_c
    float* out  = (float*)d_out;
    float* impF = out + 1 + B_SZ;           // imputations slot of d_out

    ScanArgs A;
    A.x = x; A.mask = mask; A.ts = ts; A.rec = rec;
    A.impF = impF; A.impB = impB; A.err = accp;
    A.predF = accp + 64; A.predB = accp + 256;

    for (int d = 0; d < 2; ++d) {
        int pb = 4 + d * 15; // params_f at 4..18, params_b at 19..33
        A.b_gh[d]   = (const float*)d_in[pb + 1];
        A.b_gx[d]   = (const float*)d_in[pb + 3];
        A.b_hist[d] = (const float*)d_in[pb + 5];
        A.b_feat[d] = (const float*)d_in[pb + 7];
        A.b_beta[d] = (const float*)d_in[pb + 9];
        A.b_lstm[d] = (const float*)d_in[pb + 12];
        A.w_out[d]  = (const float*)d_in[pb + 13];
        A.b_out[d]  = (const float*)d_in[pb + 14];
        unsigned short* sw = (unsigned short*)(ws + OFF_SWZ + (size_t)d * SWZ_STRIDE);
        A.whist[d] = sw;
        A.wfeat[d] = sw + 8192;
        A.wbeta[d] = sw + 12288;
        A.wih[d]   = sw + 20480;
        A.whh[d]   = sw + 86016;
        float* csp = (float*)(ws + OFF_CS + (size_t)d * 1024);
        A.cs_gh[d] = csp;
        A.cs_gx[d] = csp + 128;
    }

    brits_zero<<<1, 256, 0, stream>>>(accp);

    for (int d = 0; d < 2; ++d) {
        int pb = 4 + d * 15;
        brits_cs<<<1, 128, 0, stream>>>((const float*)d_in[pb + 0], (const float*)d_in[pb + 2],
                                        (float*)A.cs_gh[d], (float*)A.cs_gx[d]);
        brits_swz<<<32, 256, 0, stream>>>((const float*)d_in[pb + 4],
                                          (unsigned short*)A.whist[d], 128, 64, 0);
        brits_swz<<<16, 256, 0, stream>>>((const float*)d_in[pb + 6],
                                          (unsigned short*)A.wfeat[d], 64, 64, 1);
        brits_swz<<<32, 256, 0, stream>>>((const float*)d_in[pb + 8],
                                          (unsigned short*)A.wbeta[d], 128, 64, 0);
        brits_swz<<<256, 256, 0, stream>>>((const float*)d_in[pb + 10],
                                           (unsigned short*)A.wih[d], 128, 512, 0);
        brits_swz<<<256, 256, 0, stream>>>((const float*)d_in[pb + 11],
                                           (unsigned short*)A.whh[d], 128, 512, 0);
    }

    brits_den<<<1024, NTHR, 0, stream>>>(mask, rec, accp + 2);

    brits_scan<<<16, NTHR, 0, stream>>>(A);

    brits_combine<<<2048, NTHR, 0, stream>>>(impF, impB, accp + 3);

    brits_final<<<1, 128, 0, stream>>>(accp, A.predF, A.predB, out);
}
